// ConditionalFilterLayer_68367289417849
// MI455X (gfx1250) — matrix operand order repaired and checked by reading, not compiled, not run on hardware
//
#include <hip/hip_runtime.h>
#include <hip/hip_bf16.h>
#include <math.h>

typedef __attribute__((ext_vector_type(16))) __bf16 v16bf;
typedef __attribute__((ext_vector_type(8)))  __bf16 v8bf;
typedef __attribute__((ext_vector_type(8)))  float  v8f;

constexpr int Bn = 8, Cc = 512, Kk = 64, HW = 16384;
constexpr int TILE1 = 128, CHUNK1 = 4;       // kernel1: 128-px tiles, 4 per block
constexpr int TILE3 = 256;                    // kernel3: 256-px tiles
constexpr int XT_STR = 520;                   // [hw][c] row stride (pad 8 -> bank stagger)
constexpr int XC_STR = 136;                   // [c][hw] row stride (pad 8)
constexpr int AUX_STR = 136;

__device__ __forceinline__ __bf16 bf(float x) { return (__bf16)x; }

// fast sigmoid: v_exp_f32 + v_rcp_f32 (avoid IEEE divide expansion)
__device__ __forceinline__ float fast_sigmoid(float t) {
  return __builtin_amdgcn_rcpf(1.0f + __expf(-t));
}

// A fragment (16x32 bf16) from row-major f32 global: lane r=lane&15 holds row m0+r,
// elements 0..7 = K {k0+half*8..+7}, elements 8..15 = K {k0+16+half*8..+7}
__device__ __forceinline__ v16bf load_a_g(const float* __restrict__ A, int m0, int k0,
                                          int ld, int lane) {
  int r = lane & 15, half = lane >> 4;
  const float* p = A + (size_t)(m0 + r) * ld + k0 + half * 8;
  float4 f0 = *(const float4*)(p);
  float4 f1 = *(const float4*)(p + 4);
  float4 f2 = *(const float4*)(p + 16);
  float4 f3 = *(const float4*)(p + 20);
  v16bf o;
  o[0] = bf(f0.x);  o[1] = bf(f0.y);  o[2] = bf(f0.z);  o[3] = bf(f0.w);
  o[4] = bf(f1.x);  o[5] = bf(f1.y);  o[6] = bf(f1.z);  o[7] = bf(f1.w);
  o[8] = bf(f2.x);  o[9] = bf(f2.y);  o[10] = bf(f2.z); o[11] = bf(f2.w);
  o[12] = bf(f3.x); o[13] = bf(f3.y); o[14] = bf(f3.z); o[15] = bf(f3.w);
  return o;
}

// A fragment from row-major bf16 LDS
__device__ __forceinline__ v16bf load_a_s(const __bf16* A, int ld, int m0, int k0, int lane) {
  int r = lane & 15, half = lane >> 4;
  const __bf16* p = A + (m0 + r) * ld + k0 + half * 8;
  v8bf lo = *(const v8bf*)p;
  v8bf hi = *(const v8bf*)(p + 16);
  v16bf o;
#pragma unroll
  for (int i = 0; i < 8; ++i) { o[i] = lo[i]; o[8 + i] = hi[i]; }
  return o;
}

// B fragment (32x16 bf16) from K-contiguous storage Bt[n][k] (row stride ld):
// lane col n0+(lane&15); elements 0..7 = K {k0+half*8..+7}, elements 8..15 = K {k0+16+half*8..+7}
__device__ __forceinline__ v16bf load_b_s(const __bf16* Bt, int ld, int n0, int k0, int lane) {
  const __bf16* p = Bt + (n0 + (lane & 15)) * ld + k0 + ((lane >> 4) << 3);
  v8bf lo = *(const v8bf*)p;
  v8bf hi = *(const v8bf*)(p + 16);
  v16bf o;
#pragma unroll
  for (int i = 0; i < 8; ++i) { o[i] = lo[i]; o[8 + i] = hi[i]; }
  return o;
}

// ---------------- Kernel 1: fused mask-conv + sigmoid + masked pooling ----------------
__global__ void __launch_bounds__(512) fused_mask_pool(
    const float* __restrict__ x, const float* __restrict__ mask_w,
    const float* __restrict__ mask_b, float* __restrict__ aux_out,
    float* __restrict__ class_feat) {
  extern __shared__ char smem[];
  __bf16* xt  = (__bf16*)smem;                   // [128][520] : x^T, c contiguous
  __bf16* xc  = (__bf16*)(smem + 133120);        // [512][136] : x,   hw contiguous
  __bf16* aux = (__bf16*)(smem + 272384);        // [64][136]  : sigmoid tile, hw contiguous

  const int b   = blockIdx.x >> 5;               // 32 blocks per batch sample
  const int hw0 = (blockIdx.x & 31) * (TILE1 * CHUNK1);
  const int tid = threadIdx.x;
  const int lane = tid & 31, wave = tid >> 5;
  const int half = lane >> 4, lolane = lane & 15;

  v8f acc2[8];
#pragma unroll
  for (int j = 0; j < 8; ++j) acc2[j] = v8f{0, 0, 0, 0, 0, 0, 0, 0};

  const float* xb_base = x + (size_t)b * Cc * HW;

  for (int ch = 0; ch < CHUNK1; ++ch) {
    const int hwb = hw0 + ch * TILE1;
    __syncthreads();
    // ---- stage x tile into both LDS layouts (f32 -> bf16) ----
#pragma unroll 4
    for (int it = 0; it < 32; ++it) {
      int idx = it * 512 + tid;
      int hw4 = idx & 31, c = idx >> 5;
      float4 v = *(const float4*)(xb_base + (size_t)c * HW + hwb + hw4 * 4);
      __bf16 e0 = bf(v.x), e1 = bf(v.y), e2 = bf(v.z), e3 = bf(v.w);
      __bf16* pc = xc + c * XC_STR + hw4 * 4;
      pc[0] = e0; pc[1] = e1; pc[2] = e2; pc[3] = e3;
      int hwl = hw4 * 4;
      xt[(hwl + 0) * XT_STR + c] = e0;
      xt[(hwl + 1) * XT_STR + c] = e1;
      xt[(hwl + 2) * XT_STR + c] = e2;
      xt[(hwl + 3) * XT_STR + c] = e3;
    }
    __syncthreads();

    // ---- GEMM1: aux[64 x 128] = sigmoid(mask_w[64x512] @ xtile) ----
#pragma unroll
    for (int t = 0; t < 2; ++t) {
      int tt = wave + 16 * t;
      int m0 = (tt & 3) * 16, n0 = (tt >> 2) * 16;
      v8f c1 = {0, 0, 0, 0, 0, 0, 0, 0};
      for (int k0 = 0; k0 < Cc; k0 += 32) {
        v16bf a  = load_a_g(mask_w, m0, k0, Cc, lane);
        v16bf bb = load_b_s(xt, XT_STR, n0, k0, lane);   // B[n=hw][k=c]
        c1 = __builtin_amdgcn_wmma_f32_16x16x32_bf16(false, a, false, bb,
                                                     (short)0, c1, false, false);
      }
      int n = n0 + lolane;
      size_t obase = (size_t)b * Kk * HW;
#pragma unroll
      for (int r = 0; r < 8; ++r) {
        int m = m0 + half * 8 + r;
        float s = fast_sigmoid(c1[r] + mask_b[m]);
        aux_out[obase + (size_t)m * HW + hwb + n] = s;
        aux[m * AUX_STR + n] = bf(s);
      }
    }
    __syncthreads();

    // ---- GEMM2: class_feat[64 x 512] += aux[64 x 128] @ x^T ----
    {
      int m0 = (wave & 3) * 16;
      v16bf af[4];
#pragma unroll
      for (int ks = 0; ks < 4; ++ks) af[ks] = load_a_s(aux, AUX_STR, m0, ks * 32, lane);
#pragma unroll
      for (int j = 0; j < 8; ++j) {
        int n0 = ((wave >> 2) + 4 * j) * 16;
#pragma unroll
        for (int ks = 0; ks < 4; ++ks) {
          v16bf bb = load_b_s(xc, XC_STR, n0, ks * 32, lane);  // B[n=c][k=hw]
          acc2[j] = __builtin_amdgcn_wmma_f32_16x16x32_bf16(false, af[ks], false, bb,
                                                            (short)0, acc2[j], false, false);
        }
      }
    }
  }

  // ---- flush partial class_feat (scaled by 1/HW) via f32 global atomics ----
  {
    int m0 = (wave & 3) * 16;
#pragma unroll
    for (int j = 0; j < 8; ++j) {
      int cidx = ((wave >> 2) + 4 * j) * 16 + lolane;
#pragma unroll
      for (int r = 0; r < 8; ++r) {
        int m = m0 + half * 8 + r;
        unsafeAtomicAdd(&class_feat[((size_t)b * Kk + m) * Cc + cidx],
                        acc2[j][r] * (1.0f / (float)HW));
      }
    }
  }
}

// ---------------- Kernel 2: per-class filter generation (grouped GEMV) ----------------
__global__ void __launch_bounds__(256) filter_gemv(
    const float* __restrict__ filt_w, const float* __restrict__ filt_b,
    const float* __restrict__ class_feat, float* __restrict__ filters) {
  __shared__ float cf[8 * 512];
  const int k = blockIdx.x >> 6;
  const int d = ((blockIdx.x & 63) << 3) + (threadIdx.x >> 5);
  const int lane = threadIdx.x & 31;
  for (int i = threadIdx.x; i < 8 * 512; i += 256) {
    int bb = i >> 9, c = i & 511;
    cf[i] = class_feat[((size_t)bb * Kk + k) * Cc + c];
  }
  __syncthreads();
  float acc[8] = {0, 0, 0, 0, 0, 0, 0, 0};
  const float* wrow = filt_w + ((size_t)k * Cc + d) * Cc;
  for (int it = 0; it < 16; ++it) {
    int c = lane + it * 32;
    float w = wrow[c];
#pragma unroll
    for (int bb = 0; bb < 8; ++bb) acc[bb] += w * cf[(bb << 9) + c];
  }
#pragma unroll
  for (int bb = 0; bb < 8; ++bb)
    for (int off = 16; off; off >>= 1) acc[bb] += __shfl_xor(acc[bb], off, 32);
  if (lane == 0) {
    float fb = filt_b[(size_t)k * Cc + d];
#pragma unroll
    for (int bb = 0; bb < 8; ++bb)
      filters[((size_t)bb * Kk + k) * Cc + d] = acc[bb] + fb;
  }
}

// ---------------- Kernel 3: pred = filters @ x ----------------
__global__ void __launch_bounds__(512) pred_gemm(
    const float* __restrict__ x, const float* __restrict__ filters,
    float* __restrict__ pred_out) {
  extern __shared__ char smem[];
  __bf16* xt = (__bf16*)smem;                    // [256][520]
  const int b   = blockIdx.x >> 6;
  const int hwb = (blockIdx.x & 63) * TILE3;
  const int tid = threadIdx.x;
  const int lane = tid & 31, wave = tid >> 5;
  const int half = lane >> 4, lolane = lane & 15;

  const float* xb = x + (size_t)b * Cc * HW + hwb;
#pragma unroll 4
  for (int it = 0; it < 64; ++it) {
    int idx = it * 512 + tid;
    int hw4 = idx & 63, c = idx >> 6;
    float4 v = *(const float4*)(xb + (size_t)c * HW + hw4 * 4);
    int hwl = hw4 * 4;
    xt[(hwl + 0) * XT_STR + c] = bf(v.x);
    xt[(hwl + 1) * XT_STR + c] = bf(v.y);
    xt[(hwl + 2) * XT_STR + c] = bf(v.z);
    xt[(hwl + 3) * XT_STR + c] = bf(v.w);
  }
  __syncthreads();

  const float* Ab = filters + (size_t)b * Kk * Cc;
#pragma unroll
  for (int t = 0; t < 4; ++t) {
    int tt = wave + 16 * t;
    int m0 = (tt & 3) * 16, n0 = (tt >> 2) * 16;
    v8f c1 = {0, 0, 0, 0, 0, 0, 0, 0};
    for (int k0 = 0; k0 < Cc; k0 += 32) {
      v16bf a  = load_a_g(Ab, m0, k0, Cc, lane);
      v16bf bb = load_b_s(xt, XT_STR, n0, k0, lane);
      c1 = __builtin_amdgcn_wmma_f32_16x16x32_bf16(false, a, false, bb,
                                                   (short)0, c1, false, false);
    }
    int n = hwb + n0 + lolane;
#pragma unroll
    for (int r = 0; r < 8; ++r) {
      int m = m0 + half * 8 + r;
      pred_out[((size_t)b * Kk + m) * HW + n] = c1[r];
    }
  }
}

extern "C" void kernel_launch(void* const* d_in, const int* in_sizes, int n_in,
                              void* d_out, int out_size, void* d_ws, size_t ws_size,
                              hipStream_t stream) {
  const float* x      = (const float*)d_in[0];
  const float* mask_w = (const float*)d_in[1];
  const float* mask_b = (const float*)d_in[2];
  const float* filt_w = (const float*)d_in[3];
  const float* filt_b = (const float*)d_in[4];

  float* pred = (float*)d_out;
  float* aux  = pred + (size_t)Bn * Kk * HW;

  float* class_feat = (float*)d_ws;                       // [8][64][512] f32 = 1 MiB
  float* filters    = class_feat + (size_t)Bn * Kk * Cc;  // [8][64][512] f32 = 1 MiB

  const size_t lds1 = (size_t)TILE1 * XT_STR * 2 + (size_t)Cc * XC_STR * 2 +
                      (size_t)Kk * AUX_STR * 2;            // 289,792 B
  const size_t lds3 = (size_t)TILE3 * XT_STR * 2;          // 266,240 B
  hipFuncSetAttribute((const void*)fused_mask_pool,
                      hipFuncAttributeMaxDynamicSharedMemorySize, (int)lds1);
  hipFuncSetAttribute((const void*)pred_gemm,
                      hipFuncAttributeMaxDynamicSharedMemorySize, (int)lds3);

  hipMemsetAsync(class_feat, 0, (size_t)Bn * Kk * Cc * sizeof(float), stream);

  fused_mask_pool<<<Bn * 32, 512, lds1, stream>>>(x, mask_w, mask_b, aux, class_feat);
  filter_gemv<<<Kk * 64, 256, 0, stream>>>(filt_w, filt_b, class_feat, filters);
  pred_gemm<<<Bn * 64, 512, lds3, stream>>>(x, filters, pred);
}
